// MemoryEfficientMoE_29884382446041
// MI455X (gfx1250) — matrix operand order repaired and checked by reading, not compiled, not run on hardware
//
#include <hip/hip_runtime.h>
#include <hip/hip_bf16.h>

// ---- problem dims (match reference) ----
#define NTOK   16384          // B*T
#define CDIM   1024
#define EXP    8
#define HDIM   4096
#define KN     (2 * NTOK)
#define CAPACITY 2458         // int(1.2 * (NTOK/EXP + 1))
#define CAP_PAD  2560         // round up to 128-row tiles

// ---- CDNA5 WMMA types ----
typedef __bf16 bf16_t;
typedef bf16_t v16bf __attribute__((ext_vector_type(16)));
typedef float  v8f   __attribute__((ext_vector_type(8)));
typedef unsigned int u32x4 __attribute__((ext_vector_type(4)));
typedef int          i32x4 __attribute__((ext_vector_type(4)));
typedef int          i32x8 __attribute__((ext_vector_type(8)));

#if __has_builtin(__builtin_amdgcn_tensor_load_to_lds) && __has_builtin(__builtin_amdgcn_s_wait_tensorcnt)
#define USE_TDM 1
#else
#define USE_TDM 0
#endif

__device__ __forceinline__ unsigned short f2bf(float f) {
  union { float f; unsigned u; } t; t.f = f;
  unsigned r = t.u + 0x7fffu + ((t.u >> 16) & 1u);   // round-to-nearest-even
  return (unsigned short)(r >> 16);
}

// LDS tile pitch: 32 K-elements padded to 40 (80B) -> conflict-free b128 reads
#define LPITCH 40

#if USE_TDM
// One TDM descriptor loads a 128x32 bf16 tile (row pitch `stride` elems) into
// LDS at `lds_off`, inserting 16B padding after every 64B row (-> LPITCH=40).
__device__ __forceinline__ void tdm_load_tile(unsigned lds_off,
                                              const unsigned short* g,
                                              unsigned stride_elems) {
  const unsigned long long ga = (unsigned long long)(uintptr_t)g;
  u32x4 g0;
  g0.x = 1u;                                        // count=1 valid descriptor
  g0.y = lds_off;                                   // LDS byte address
  g0.z = (unsigned)(ga & 0xffffffffu);              // global_addr[31:0]
  g0.w = (unsigned)((ga >> 32) & 0x01ffffffu)       // global_addr[56:32]
       | (2u << 30);                                // type = 2 ("image")
  i32x8 g1;
  g1[0] = (1 << 16)                                 // data_size = 2 bytes
        | (1 << 20)                                 // pad_enable
        | (3 << 22)                                 // pad_interval: 16 DWORDs (64B)
        | (3 << 25);                                // pad_amount:    4 DWORDs (16B)
  g1[1] = 0;                                        // abar addr / tensor_dim0 lo16 (2^30 -> 0)
  g1[2] = 0x4000;                                   // tensor_dim0 hi16 = 2^30, dim1 lo16 = 0
  g1[3] = (32 << 16) | 0x4000;                      // tensor_dim1 hi16 = 2^30, tile_dim0 = 32
  g1[4] = 128;                                      // tile_dim1 = 128, tile_dim2 = 0
  g1[5] = (int)stride_elems;                        // tensor_dim0_stride[31:0]
  g1[6] = 0;                                        // stride hi / dim1_stride lo
  g1[7] = 0;
  const i32x4 gz4 = {0, 0, 0, 0};                   // D# groups 2/3 unused (2D tile)
  const i32x8 gz8 = {0, 0, 0, 0, 0, 0, 0, 0};
  __builtin_amdgcn_tensor_load_to_lds(g0, g1, gz4, gz4, gz8, 0);
}
#endif

// ============================= router =============================
__global__ __launch_bounds__(256) void moe_router(
    const float* __restrict__ x, const float* __restrict__ rw,
    const float* __restrict__ rb, int* __restrict__ eidx,
    float* __restrict__ ew) {
  const int wave = threadIdx.x >> 5, lane = threadIdx.x & 31;
  const int n = blockIdx.x * 8 + wave;
  const float* xr = x + (size_t)n * CDIM;
  float acc[EXP];
#pragma unroll
  for (int e = 0; e < EXP; ++e) acc[e] = 0.f;
  for (int c = lane; c < CDIM; c += 32) {
    const float xv = xr[c];
#pragma unroll
    for (int e = 0; e < EXP; ++e) acc[e] += xv * rw[e * CDIM + c];
  }
#pragma unroll
  for (int e = 0; e < EXP; ++e)
#pragma unroll
    for (int off = 16; off > 0; off >>= 1)
      acc[e] += __shfl_xor(acc[e], off, 32);

  if (lane == 0) {
    float lg[EXP], m = -1e30f;
#pragma unroll
    for (int e = 0; e < EXP; ++e) { lg[e] = acc[e] + rb[e]; m = fmaxf(m, lg[e]); }
    float s = 0.f;
#pragma unroll
    for (int e = 0; e < EXP; ++e) { lg[e] = __expf(lg[e] - m); s += lg[e]; }
    const float inv = 1.f / s;
    int i0 = 0; float v0 = -1.f;
#pragma unroll
    for (int e = 0; e < EXP; ++e) { float p = lg[e] * inv; if (p > v0) { v0 = p; i0 = e; } }
    int i1 = 0; float v1 = -1.f;
#pragma unroll
    for (int e = 0; e < EXP; ++e) { float p = lg[e] * inv; if (e != i0 && p > v1) { v1 = p; i1 = e; } }
    const float denom = v0 + v1 + 1e-9f;
    eidx[n] = i0;        ew[n] = v0 / denom;
    eidx[NTOK + n] = i1; ew[NTOK + n] = v1 / denom;
  }
}

// ============================ assignment ===========================
__global__ void moe_assign(const int* __restrict__ eidx, const float* __restrict__ ew,
                           int* __restrict__ tok, float* __restrict__ wgt,
                           int* __restrict__ cnt) {
  const int e = blockIdx.x, lane = threadIdx.x;
  int base = 0;
  for (int s0 = 0; s0 < KN; s0 += 32) {
    const int s = s0 + lane;
    const float w = ew[s];
    const bool match = (eidx[s] == e) && (w > 1e-9f);
#if __has_builtin(__builtin_amdgcn_ballot_w32)
    const unsigned mask = __builtin_amdgcn_ballot_w32(match);
#else
    const unsigned mask = (unsigned)__ballot(match);
#endif
    const int pos = base + __popc(mask & ((1u << lane) - 1u));
    if (match && pos < CAPACITY) {
      tok[e * CAP_PAD + pos] = (s >= NTOK) ? (s - NTOK) : s;   // token = slot % N
      wgt[e * CAP_PAD + pos] = w;
    }
    base += __popc(mask);
  }
  const int kept = base < CAPACITY ? base : CAPACITY;
  if (lane == 0) cnt[e] = kept;
  for (int r = kept + lane; r < CAP_PAD; r += 32) {
    tok[e * CAP_PAD + r] = 0;
    wgt[e * CAP_PAD + r] = 0.f;
  }
}

// ==================== fp32 -> bf16 bulk conversion ====================
__global__ __launch_bounds__(256) void moe_cvt_bf16(
    const float* __restrict__ src, unsigned short* __restrict__ dst) {
  const size_t idx = ((size_t)blockIdx.x * 256 + threadIdx.x) * 4;
  const float4 v = *(const float4*)(src + idx);
  ushort4 o; o.x = f2bf(v.x); o.y = f2bf(v.y); o.z = f2bf(v.z); o.w = f2bf(v.w);
  *(ushort4*)(dst + idx) = o;
}

// ======================= gather + bf16 convert ======================
__global__ __launch_bounds__(256) void moe_gather(
    const float* __restrict__ x, const int* __restrict__ tok,
    unsigned short* __restrict__ A) {
  const size_t idx = ((size_t)blockIdx.x * 256 + threadIdx.x) * 4;
  const size_t row = idx / CDIM;
  const int c = (int)(idx % CDIM);
  const int t = tok[row];
  const float4 v = *(const float4*)(x + (size_t)t * CDIM + c);
  ushort4 o; o.x = f2bf(v.x); o.y = f2bf(v.y); o.z = f2bf(v.z); o.w = f2bf(v.w);
  *(ushort4*)(A + idx) = o;
}

// ---- WMMA fragment loads from LDS (ISA 7.12.2 layouts) ----
__device__ __forceinline__ v16bf fragA(const unsigned short* rb, int kh) {
  union { uint4 u[2]; v16bf v; } t;
  t.u[0] = *(const uint4*)(rb + kh * 8);        // K 0-7  | 8-15
  t.u[1] = *(const uint4*)(rb + kh * 8 + 16);   // K 16-23| 24-31
  return t.v;
}
__device__ __forceinline__ v16bf fragB(const unsigned short* rb, int kh) {
  union { uint4 u[2]; v16bf v; } t;
  t.u[0] = *(const uint4*)(rb + kh * 8);        // K 0-7  | 8-15
  t.u[1] = *(const uint4*)(rb + kh * 8 + 16);   // K 16-23| 24-31
  return t.v;
}

// ===================== GEMM1: hidden = gelu(A @ W1^T + b1) =====================
__global__ __launch_bounds__(256) void moe_fc1(
    const unsigned short* __restrict__ Abf, const unsigned short* __restrict__ w1,
    const float* __restrict__ b1, unsigned short* __restrict__ hid) {
  __shared__ __align__(16) unsigned short lA[2][128 * LPITCH];
  __shared__ __align__(16) unsigned short lB[2][128 * LPITCH];
  const int e = blockIdx.z;
  const int h0 = blockIdx.x * 128;
  const int m0 = blockIdx.y * 128;
  const int lane = threadIdx.x & 31;
  const int wave = threadIdx.x >> 5;
  const int wm = wave >> 2, wn = wave & 3;
  const unsigned short* Ae = Abf + (size_t)e * CAP_PAD * CDIM + (size_t)m0 * CDIM;
  const unsigned short* We = w1 + (size_t)e * HDIM * CDIM + (size_t)h0 * CDIM;

  const v8f vz = {0.f, 0.f, 0.f, 0.f, 0.f, 0.f, 0.f, 0.f};
  v8f acc[4][2];
#pragma unroll
  for (int mf = 0; mf < 4; ++mf)
#pragma unroll
    for (int nf = 0; nf < 2; ++nf) acc[mf][nf] = vz;

  auto issueTiles = [&](int kt, int buf) {
    const int k0 = kt * 32;
#if USE_TDM
    if (wave == 0) {       // one DMA descriptor per tile, TENSORcnt tracked
      tdm_load_tile((unsigned)(uintptr_t)&lA[buf][0], Ae + k0, CDIM);
      tdm_load_tile((unsigned)(uintptr_t)&lB[buf][0], We + k0, CDIM);
    }
#else
#pragma unroll
    for (int i = 0; i < 2; ++i) {
      const int q = threadIdx.x + i * 256;
      const int row = q >> 2, col = q & 3;
      *(uint4*)&lA[buf][row * LPITCH + col * 8] =
          *(const uint4*)(Ae + (size_t)row * CDIM + k0 + col * 8);
    }
#pragma unroll
    for (int i = 0; i < 2; ++i) {
      const int q = threadIdx.x + i * 256;
      const int row = q >> 2, col = q & 3;
      *(uint4*)&lB[buf][row * LPITCH + col * 8] =
          *(const uint4*)(We + (size_t)row * CDIM + k0 + col * 8);
    }
#endif
  };
  auto waitTiles = [&]() {
#if USE_TDM
    if (wave == 0) __builtin_amdgcn_s_wait_tensorcnt(0);
#endif
  };

  auto doMMA = [&](int buf) {
    const int l15 = lane & 15, kh = lane >> 4;
    v16bf afr[4], bfr[2];
#pragma unroll
    for (int mf = 0; mf < 4; ++mf)
      afr[mf] = fragA(&lA[buf][(wm * 64 + mf * 16 + l15) * LPITCH], kh);
#pragma unroll
    for (int nf = 0; nf < 2; ++nf)
      bfr[nf] = fragB(&lB[buf][(wn * 32 + nf * 16 + l15) * LPITCH], kh);
#pragma unroll
    for (int mf = 0; mf < 4; ++mf)
#pragma unroll
      for (int nf = 0; nf < 2; ++nf)
        acc[mf][nf] = __builtin_amdgcn_wmma_f32_16x16x32_bf16(
            false, afr[mf], false, bfr[nf], (short)0, acc[mf][nf], false, false);
  };

  const int NK = CDIM / 32;
  issueTiles(0, 0);
  waitTiles();
  __syncthreads();
  for (int kt = 0; kt < NK; ++kt) {
    if (kt + 1 < NK) issueTiles(kt + 1, (kt + 1) & 1);
    doMMA(kt & 1);
    if (kt + 1 < NK) waitTiles();
    __syncthreads();
  }

#pragma unroll
  for (int nf = 0; nf < 2; ++nf) {
    const int h = h0 + wn * 32 + nf * 16 + (lane & 15);
    const float bias = b1[e * HDIM + h];
#pragma unroll
    for (int mf = 0; mf < 4; ++mf)
#pragma unroll
      for (int i = 0; i < 8; ++i) {
        const int m = m0 + wm * 64 + mf * 16 + i + (lane >> 4) * 8;
        const float v = acc[mf][nf][i] + bias;
        const float g = 0.5f * v * (1.f + erff(v * 0.70710678118654752f));
        hid[((size_t)e * CAP_PAD + m) * HDIM + h] = f2bf(g);
      }
  }
}

// ============ GEMM2: y = hid @ W2^T + b2; fused residual/scale/scatter ============
__global__ __launch_bounds__(256) void moe_fc2(
    const unsigned short* __restrict__ hid, const unsigned short* __restrict__ w2,
    const float* __restrict__ b2, const float* __restrict__ lscale,
    const float* __restrict__ x, const int* __restrict__ tok,
    const float* __restrict__ wgt, float* __restrict__ out) {
  __shared__ __align__(16) unsigned short lA[2][128 * LPITCH];
  __shared__ __align__(16) unsigned short lB[2][128 * LPITCH];
  const int e = blockIdx.z;
  const int c0 = blockIdx.x * 128;
  const int m0 = blockIdx.y * 128;
  const int lane = threadIdx.x & 31;
  const int wave = threadIdx.x >> 5;
  const int wm = wave >> 2, wn = wave & 3;
  const unsigned short* Ae = hid + (size_t)e * CAP_PAD * HDIM + (size_t)m0 * HDIM;
  const unsigned short* We = w2 + (size_t)e * CDIM * HDIM + (size_t)c0 * HDIM;

  const v8f vz = {0.f, 0.f, 0.f, 0.f, 0.f, 0.f, 0.f, 0.f};
  v8f acc[4][2];
#pragma unroll
  for (int mf = 0; mf < 4; ++mf)
#pragma unroll
    for (int nf = 0; nf < 2; ++nf) acc[mf][nf] = vz;

  auto issueTiles = [&](int kt, int buf) {
    const int k0 = kt * 32;
#if USE_TDM
    if (wave == 0) {
      tdm_load_tile((unsigned)(uintptr_t)&lA[buf][0], Ae + k0, HDIM);
      tdm_load_tile((unsigned)(uintptr_t)&lB[buf][0], We + k0, HDIM);
    }
#else
#pragma unroll
    for (int i = 0; i < 2; ++i) {
      const int q = threadIdx.x + i * 256;
      const int row = q >> 2, col = q & 3;
      *(uint4*)&lA[buf][row * LPITCH + col * 8] =
          *(const uint4*)(Ae + (size_t)row * HDIM + k0 + col * 8);
    }
#pragma unroll
    for (int i = 0; i < 2; ++i) {
      const int q = threadIdx.x + i * 256;
      const int row = q >> 2, col = q & 3;
      *(uint4*)&lB[buf][row * LPITCH + col * 8] =
          *(const uint4*)(We + (size_t)row * HDIM + k0 + col * 8);
    }
#endif
  };
  auto waitTiles = [&]() {
#if USE_TDM
    if (wave == 0) __builtin_amdgcn_s_wait_tensorcnt(0);
#endif
  };

  auto doMMA = [&](int buf) {
    const int l15 = lane & 15, kh = lane >> 4;
    v16bf afr[4], bfr[2];
#pragma unroll
    for (int mf = 0; mf < 4; ++mf)
      afr[mf] = fragA(&lA[buf][(wm * 64 + mf * 16 + l15) * LPITCH], kh);
#pragma unroll
    for (int nf = 0; nf < 2; ++nf)
      bfr[nf] = fragB(&lB[buf][(wn * 32 + nf * 16 + l15) * LPITCH], kh);
#pragma unroll
    for (int mf = 0; mf < 4; ++mf)
#pragma unroll
      for (int nf = 0; nf < 2; ++nf)
        acc[mf][nf] = __builtin_amdgcn_wmma_f32_16x16x32_bf16(
            false, afr[mf], false, bfr[nf], (short)0, acc[mf][nf], false, false);
  };

  const int NK = HDIM / 32;
  issueTiles(0, 0);
  waitTiles();
  __syncthreads();
  for (int kt = 0; kt < NK; ++kt) {
    if (kt + 1 < NK) issueTiles(kt + 1, (kt + 1) & 1);
    doMMA(kt & 1);
    if (kt + 1 < NK) waitTiles();
    __syncthreads();
  }

#pragma unroll
  for (int nf = 0; nf < 2; ++nf) {
    const int c = c0 + wn * 32 + nf * 16 + (lane & 15);
    const float bias = b2[e * CDIM + c];
    const float sc = lscale[e * CDIM + c];
#pragma unroll
    for (int mf = 0; mf < 4; ++mf)
#pragma unroll
      for (int i = 0; i < 8; ++i) {
        const int m = m0 + wm * 64 + mf * 16 + i + (lane >> 4) * 8;
        const int t = tok[e * CAP_PAD + m];
        const float w = wgt[e * CAP_PAD + m];
        if (w > 0.f) {
          const float y = acc[mf][nf][i] + bias;
          const float r = x[(size_t)t * CDIM + c] + y * sc;
          atomicAdd(out + (size_t)t * CDIM + c, w * r);
        }
      }
  }
}

// =============================== launch ===============================
extern "C" void kernel_launch(void* const* d_in, const int* in_sizes, int n_in,
                              void* d_out, int out_size, void* d_ws, size_t ws_size,
                              hipStream_t stream) {
  const float* x        = (const float*)d_in[0];
  const float* router_w = (const float*)d_in[1];
  const float* router_b = (const float*)d_in[2];
  const float* fc1_w    = (const float*)d_in[3];
  const float* fc1_b    = (const float*)d_in[4];
  const float* fc2_w    = (const float*)d_in[5];
  const float* fc2_b    = (const float*)d_in[6];
  const float* lscale   = (const float*)d_in[7];
  float* out = (float*)d_out;

  char* ws = (char*)d_ws;
  int*   eidx = (int*)(ws + 0);                          // KN ints       (128 KB)
  float* ew   = (float*)(ws + 131072);                   // KN floats     (128 KB)
  int*   tok  = (int*)(ws + 262144);                     // E*CAP_PAD     ( 80 KB)
  float* wgt  = (float*)(ws + 344064);                   // E*CAP_PAD     ( 80 KB)
  int*   cnt  = (int*)(ws + 425984);
  unsigned short* w1bf = (unsigned short*)(ws + 430080);            // 64 MB
  unsigned short* w2bf = (unsigned short*)(ws + 67538944);          // 64 MB
  unsigned short* Abf  = (unsigned short*)(ws + 134647808);         // 40 MB
  unsigned short* hid  = (unsigned short*)(ws + 176590848);         // 160 MB

  (void)hipMemsetAsync(d_out, 0, (size_t)NTOK * CDIM * sizeof(float), stream);

  // one-shot weight conversion to bf16 (enables pure-copy TDM tile loads)
  moe_cvt_bf16<<<(EXP * HDIM * CDIM / 4) / 256, 256, 0, stream>>>(fc1_w, w1bf);
  moe_cvt_bf16<<<(EXP * CDIM * HDIM / 4) / 256, 256, 0, stream>>>(fc2_w, w2bf);

  moe_router<<<NTOK / 8, 256, 0, stream>>>(x, router_w, router_b, eidx, ew);
  moe_assign<<<EXP, 32, 0, stream>>>(eidx, ew, tok, wgt, cnt);
  moe_gather<<<(EXP * CAP_PAD * CDIM / 4) / 256, 256, 0, stream>>>(x, tok, Abf);

  dim3 g1(HDIM / 128, CAP_PAD / 128, EXP);   // 32 x 20 x 8
  moe_fc1<<<g1, 256, 0, stream>>>(Abf, w1bf, fc1_b, hid);

  dim3 g2(CDIM / 128, CAP_PAD / 128, EXP);   // 8 x 20 x 8
  moe_fc2<<<g2, 256, 0, stream>>>(hid, w2bf, fc2_b, lscale, x, tok, wgt, out);
}
